// GCNEncoder_39642548142599
// MI455X (gfx1250) — matrix operand order repaired and checked by reading, not compiled, not run on hardware
//
#include <hip/hip_runtime.h>
#include <hip/hip_bf16.h>
#include <math.h>

typedef __attribute__((ext_vector_type(16))) __bf16 v16bf;
typedef __attribute__((ext_vector_type(8)))  float  v8f;

#define N_NODES   10000
#define N_EDGES   160000
#define F_IN      128
#define HC        512     // H*C for both layers
#define H_HEADS   8
#define C_CH      64
#define NEG_SLOPE 0.2f
#define EPSV      1e-16f

// ---------- order-preserving float<->uint encoding for atomicMax ----------
__device__ __forceinline__ unsigned fenc(float f) {
  unsigned u = __float_as_uint(f);
  return (u & 0x80000000u) ? ~u : (u | 0x80000000u);
}
__device__ __forceinline__ float fdec(unsigned u) {
  return __uint_as_float((u & 0x80000000u) ? (u ^ 0x80000000u) : ~u);
}
// encoded 0 decodes to -NaN => "empty segment" sentinel; reference maps
// non-finite segment max to 0.0
__device__ __forceinline__ float fixmax(float m) {
  return (__builtin_fabsf(m) < __builtin_inff()) ? m : 0.0f;
}

// ---------- f32 -> bf16 conversion (row-major, for A matrices) ----------
__global__ void k_f32_to_bf16(const float* __restrict__ in, __bf16* __restrict__ out, int n) {
  int i = blockIdx.x * blockDim.x + threadIdx.x;
  if (i < n) out[i] = (__bf16)in[i];
}

// ---------- f32 [K,Nc] -> bf16 [Nc,K] transposed conversion (for B/weights) ----
// makes per-lane WMMA B fragments contiguous => global_load_b128
__global__ void k_f32_to_bf16_T(const float* __restrict__ in, __bf16* __restrict__ out,
                                int K, int Nc) {
  int i = blockIdx.x * blockDim.x + threadIdx.x;
  if (i >= K * Nc) return;
  int k = i / Nc, n = i % Nc;
  out[(size_t)n * K + k] = (__bf16)in[i];
}

// ---------- bf16 WMMA GEMM: C[M,Nc] = A[M,K] @ Bt[Nc,K]^T, f32 accum ----------
// one wave computes a 16x64 output slab (4 n-tiles): A fragment loaded once per
// k-step and reused across 4 WMMAs. K, Nc compile-time => immediate offsets.
template <int K, int Nc>
__global__ void k_gemm_bf16(const __bf16* __restrict__ A, const __bf16* __restrict__ Bt,
                            float* __restrict__ Cout) {
  constexpr int NT = 4;                         // n-tiles per wave
  constexpr int NGRP = (Nc / 16) / NT;          // slab groups per row of tiles
  int wave = blockIdx.x * (blockDim.x >> 5) + (threadIdx.x >> 5);
  int lane = threadIdx.x & 31;
  int m0 = (wave / NGRP) << 4;
  int n0 = (wave % NGRP) * (16 * NT);
  bool hi = lane >= 16;
  int  l  = lane & 15;
  // base pointers for this lane (16B aligned: K mult of 32, buffers 256B aligned)
  const __bf16* abase = A  + (size_t)(m0 + l) * K + (hi ? 8 : 0);
  const __bf16* bbase = Bt + (size_t)(n0 + l) * K + (hi ? 8 : 0);
  v8f acc[NT] = {};
  union U { v16bf v; uint4 q[2]; };
#pragma unroll 2
  for (int k0 = 0; k0 < K; k0 += 32) {
    // A 16x32 bf16 layout (ISA 7.12.2): lane<16 -> K 0..7 & 16..23, lane>=16 -> +8
    U a;
    const uint4* ap = (const uint4*)(abase + k0);
    a.q[0] = ap[0];             // K k0..+7    (elems 0..7)
    a.q[1] = ap[2];             // K k0+16..23 (elems 8..15)
#pragma unroll
    for (int s = 0; s < NT; ++s) {
      // B 32x16 bf16: lane<16 -> col n0+s*16+l, K k0..7 & 16..23; lane>=16 -> +8
      U b;
      const uint4* bp = (const uint4*)(bbase + (size_t)s * 16 * K + k0);
      b.q[0] = bp[0];
      b.q[1] = bp[2];
      acc[s] = __builtin_amdgcn_wmma_f32_16x16x32_bf16(false, a.v, false, b.v, (short)0,
                                                       acc[s], false, false);
    }
  }
  // C/D 16x16 f32: VGPR i -> row i (lanes 0-15) / row 8+i (lanes 16-31), col = +l
  float* crow = Cout + (size_t)(m0 + (hi ? 8 : 0)) * Nc + n0 + l;
#pragma unroll
  for (int s = 0; s < NT; ++s)
#pragma unroll
    for (int i = 0; i < 8; ++i) crow[i * Nc + s * 16] = acc[s][i];
}

// ---------- per-(node,head) attention coefficients ----------
__global__ void k_node_alpha(const float* __restrict__ hlin,
                             const float* __restrict__ att_s, const float* __restrict__ att_d,
                             float* __restrict__ as_, float* __restrict__ ad_) {
  int i = blockIdx.x * blockDim.x + threadIdx.x;   // n*H + h
  if (i >= N_NODES * H_HEADS) return;
  int n = i / H_HEADS, h = i % H_HEADS;
  const float* row = hlin + (size_t)n * HC + h * C_CH;
  const float* wsv = att_s + h * C_CH;
  const float* wdv = att_d + h * C_CH;
  float ss = 0.f, sd = 0.f;
  for (int c = 0; c < C_CH; ++c) { float v = row[c]; ss += v * wsv[c]; sd += v * wdv[c]; }
  as_[i] = ss; ad_[i] = sd;
}

// ---------- edge logits: leaky-relu + segment max ----------
__global__ void k_edge_logit_max(const int* __restrict__ src, const int* __restrict__ dst,
                                 const float* __restrict__ as_, const float* __restrict__ ad_,
                                 float* __restrict__ ex_e, unsigned* __restrict__ mx1) {
  int i = blockIdx.x * blockDim.x + threadIdx.x;   // e*H + h
  if (i >= N_EDGES * H_HEADS) return;
  int e = i / H_HEADS, h = i % H_HEADS;
  int s = src[e], d = dst[e];
  float v = as_[s * H_HEADS + h] + ad_[d * H_HEADS + h];
  v = v > 0.f ? v : NEG_SLOPE * v;
  ex_e[i] = v;
  atomicMax(&mx1[d * H_HEADS + h], fenc(v));
}

// ---------- edge exp + segment denominator ----------
__global__ void k_edge_exp(const int* __restrict__ dst, float* __restrict__ ex_e,
                           const unsigned* __restrict__ mx1, float* __restrict__ den1) {
  int i = blockIdx.x * blockDim.x + threadIdx.x;
  if (i >= N_EDGES * H_HEADS) return;
  int e = i / H_HEADS, h = i % H_HEADS;
  int d = dst[e];
  float mv = fixmax(fdec(mx1[d * H_HEADS + h]));
  float ex = expf(ex_e[i] - mv);
  ex_e[i] = ex;
  atomicAdd(&den1[d * H_HEADS + h], ex);
}

// ---------- message pass A: per-channel segment max of t*m ----------
// 128 threads per edge, 4 channels each (float4 gather of h[src])
__global__ void k_edge_msg_max(const int* __restrict__ src, const int* __restrict__ dst,
                               const float* __restrict__ ex_e, const float* __restrict__ den1,
                               const float* __restrict__ hlin, const float* __restrict__ tp,
                               unsigned* __restrict__ mx2) {
  int tid = blockIdx.x * blockDim.x + threadIdx.x;
  int e = tid >> 7;
  if (e >= N_EDGES) return;
  int sub = tid & 127;
  int c4 = sub << 2;               // channel base; whole float4 within one head
  int h  = c4 >> 6;
  int s = src[e], d = dst[e];
  float t = *tp;
  float a = ex_e[e * H_HEADS + h] / (den1[d * H_HEADS + h] + EPSV);
  float4 hv = *(const float4*)(hlin + (size_t)s * HC + c4);
  unsigned* mrow = mx2 + (size_t)d * HC + c4;
  atomicMax(&mrow[0], fenc(t * (hv.x * a)));
  atomicMax(&mrow[1], fenc(t * (hv.y * a)));
  atomicMax(&mrow[2], fenc(t * (hv.z * a)));
  atomicMax(&mrow[3], fenc(t * (hv.w * a)));
}

// ---------- message pass B: numerator / denominator segment sums ----------
__global__ void k_edge_msg_sum(const int* __restrict__ src, const int* __restrict__ dst,
                               const float* __restrict__ ex_e, const float* __restrict__ den1,
                               const float* __restrict__ hlin, const float* __restrict__ tp,
                               const unsigned* __restrict__ mx2,
                               float* __restrict__ den2, float* __restrict__ num2) {
  int tid = blockIdx.x * blockDim.x + threadIdx.x;
  int e = tid >> 7;
  if (e >= N_EDGES) return;
  int sub = tid & 127;
  int c4 = sub << 2;
  int h  = c4 >> 6;
  int s = src[e], d = dst[e];
  float t = *tp;
  float a = ex_e[e * H_HEADS + h] / (den1[d * H_HEADS + h] + EPSV);
  float4 hv = *(const float4*)(hlin + (size_t)s * HC + c4);
  const unsigned* mrow = mx2 + (size_t)d * HC + c4;
  float*   drow = den2 + (size_t)d * HC + c4;
  float*   nrow = num2 + (size_t)d * HC + c4;
  float m[4] = { hv.x * a, hv.y * a, hv.z * a, hv.w * a };
#pragma unroll
  for (int j = 0; j < 4; ++j) {
    float mv = fixmax(fdec(mrow[j]));
    float ex = expf(t * m[j] - mv);
    atomicAdd(&drow[j], ex);
    atomicAdd(&nrow[j], ex * m[j]);
  }
}

// ---------- finalize: out = num/(den+eps) + bias, relu ----------
__global__ void k_node_out(const float* __restrict__ num2, const float* __restrict__ den2,
                           const float* __restrict__ bias, float* __restrict__ out) {
  int i = blockIdx.x * blockDim.x + threadIdx.x;
  if (i >= N_NODES * HC) return;
  float v = num2[i] / (den2[i] + EPSV) + bias[i % HC];
  out[i] = v > 0.f ? v : 0.f;
}

extern "C" void kernel_launch(void* const* d_in, const int* in_sizes, int n_in,
                              void* d_out, int out_size, void* d_ws, size_t ws_size,
                              hipStream_t stream) {
  (void)in_sizes; (void)n_in; (void)out_size; (void)ws_size;
  const float* x    = (const float*)d_in[0];
  const int*   eidx = (const int*)d_in[1];
  const int*   src  = eidx;
  const int*   dst  = eidx + N_EDGES;
  const float* W1   = (const float*)d_in[2];
  const float* as1  = (const float*)d_in[3];
  const float* ad1  = (const float*)d_in[4];
  const float* b1   = (const float*)d_in[5];
  const float* t1   = (const float*)d_in[6];
  const float* W2   = (const float*)d_in[7];
  const float* as2  = (const float*)d_in[8];
  const float* ad2  = (const float*)d_in[9];
  const float* b2   = (const float*)d_in[10];
  const float* t2   = (const float*)d_in[11];
  float* out = (float*)d_out;

  char* ws = (char*)d_ws;
  size_t off = 0;
  auto alloc = [&](size_t bytes) { char* p = ws + off; off += (bytes + 255) & ~(size_t)255; return p; };
  __bf16*   xb   = (__bf16*)  alloc((size_t)N_NODES * F_IN * 2);
  __bf16*   w1t  = (__bf16*)  alloc((size_t)F_IN * HC * 2);      // transposed [HC, F_IN]
  __bf16*   w2t  = (__bf16*)  alloc((size_t)HC * HC * 2);        // transposed [HC, HC]
  float*    hlin = (float*)   alloc((size_t)N_NODES * HC * 4);   // current layer GEMM out
  float*    l1o  = (float*)   alloc((size_t)N_NODES * HC * 4);   // layer-1 output (f32)
  __bf16*   l1b  = (__bf16*)  alloc((size_t)N_NODES * HC * 2);   // layer-1 output (bf16)
  float*    asb  = (float*)   alloc((size_t)N_NODES * H_HEADS * 4);
  float*    adb  = (float*)   alloc((size_t)N_NODES * H_HEADS * 4);
  unsigned* mx1  = (unsigned*)alloc((size_t)N_NODES * H_HEADS * 4);
  float*    den1 = (float*)   alloc((size_t)N_NODES * H_HEADS * 4);
  float*    exe  = (float*)   alloc((size_t)N_EDGES * H_HEADS * 4);
  unsigned* mx2  = (unsigned*)alloc((size_t)N_NODES * HC * 4);
  float*    den2 = (float*)   alloc((size_t)N_NODES * HC * 4);
  float*    num2 = (float*)   alloc((size_t)N_NODES * HC * 4);

  const int TPB = 256;
  auto cdiv = [](long long a, long long b) { return (int)((a + b - 1) / b); };

  // ---- conversions (weights transposed for contiguous WMMA B fragments) ----
  k_f32_to_bf16<<<cdiv((long long)N_NODES * F_IN, TPB), TPB, 0, stream>>>(x, xb, N_NODES * F_IN);
  k_f32_to_bf16_T<<<cdiv((long long)F_IN * HC, TPB), TPB, 0, stream>>>(W1, w1t, F_IN, HC);
  k_f32_to_bf16_T<<<cdiv((long long)HC * HC, TPB), TPB, 0, stream>>>(W2, w2t, HC, HC);

  const int EH_blocks  = cdiv((long long)N_EDGES * H_HEADS, TPB);
  const int NH_blocks  = cdiv((long long)N_NODES * H_HEADS, TPB);
  const int MSG_blocks = cdiv((long long)N_EDGES * 128, TPB);
  const int NHC_blocks = cdiv((long long)N_NODES * HC, TPB);
  // 625 m-tiles x 8 slab-groups = 5000 waves, 8 waves/block -> 625 blocks (exact)
  const int GEMM_blocks = (N_NODES / 16) * ((HC / 16) / 4) / 8;

  auto attention = [&](const float* att_s, const float* att_d, const float* bias,
                       const float* tscale, float* layer_out) {
    k_node_alpha<<<NH_blocks, TPB, 0, stream>>>(hlin, att_s, att_d, asb, adb);
    hipMemsetAsync(mx1, 0, (size_t)N_NODES * H_HEADS * 4, stream);  // enc sentinel
    hipMemsetAsync(den1, 0, (size_t)N_NODES * H_HEADS * 4, stream);
    hipMemsetAsync(mx2, 0, (size_t)N_NODES * HC * 4, stream);
    hipMemsetAsync(den2, 0, (size_t)N_NODES * HC * 4, stream);
    hipMemsetAsync(num2, 0, (size_t)N_NODES * HC * 4, stream);
    k_edge_logit_max<<<EH_blocks, TPB, 0, stream>>>(src, dst, asb, adb, exe, mx1);
    k_edge_exp<<<EH_blocks, TPB, 0, stream>>>(dst, exe, mx1, den1);
    k_edge_msg_max<<<MSG_blocks, TPB, 0, stream>>>(src, dst, exe, den1, hlin, tscale, mx2);
    k_edge_msg_sum<<<MSG_blocks, TPB, 0, stream>>>(src, dst, exe, den1, hlin, tscale, mx2,
                                                   den2, num2);
    k_node_out<<<NHC_blocks, TPB, 0, stream>>>(num2, den2, bias, layer_out);
  };

  // layer 1: x(bf16) @ W1, attention-aggregate, relu -> l1o
  k_gemm_bf16<F_IN, HC><<<GEMM_blocks, 256, 0, stream>>>(xb, w1t, hlin);
  attention(as1, ad1, b1, t1, l1o);
  k_f32_to_bf16<<<NHC_blocks, TPB, 0, stream>>>(l1o, l1b, N_NODES * HC);
  // layer 2: l1(bf16) @ W2 -> final output
  k_gemm_bf16<HC, HC><<<GEMM_blocks, 256, 0, stream>>>(l1b, w2t, hlin);
  attention(as2, ad2, b2, t2, out);
}
